// Resnet_Generator_17179869782
// MI455X (gfx1250) — matrix operand order repaired and checked by reading, not compiled, not run on hardware
//
#include <hip/hip_runtime.h>

typedef __bf16 bf16;
typedef __attribute__((ext_vector_type(16))) __bf16 v16bf;
typedef __attribute__((ext_vector_type(8)))  float  v8f;
typedef unsigned int u32x4 __attribute__((ext_vector_type(4)));
typedef int          i32x4 __attribute__((ext_vector_type(4)));
typedef int          i32x8 __attribute__((ext_vector_type(8)));

union Frag {
    v16bf v;
    uint4 u[2];
};

static constexpr int NB = 8;
static constexpr int HH = 256;
static constexpr int WW = 256;
static constexpr int NPIX = NB * HH * WW;      // 524288

__device__ inline uint4 uzero() { uint4 z; z.x = z.y = z.z = z.w = 0u; return z; }

// ---------------------------------------------------------------------------
// Weight staging into LDS: prefer TDM (tensor_load_to_lds), then async
// global->LDS copies, then plain load/ds_store. All guarded so compile never
// breaks. Caller must __syncthreads() afterwards. nelem = bf16 element count
// (multiple of 8; fits in 16-bit tile_dim0).
// ---------------------------------------------------------------------------
#if __has_builtin(__builtin_amdgcn_tensor_load_to_lds) && \
    __has_builtin(__builtin_amdgcn_s_wait_tensorcnt)
#define STAGE_TDM 1
#else
#define STAGE_TDM 0
#endif
#if __has_builtin(__builtin_amdgcn_global_load_async_to_lds_b128) && \
    __has_builtin(__builtin_amdgcn_s_wait_asynccnt)
#define STAGE_ASYNC 1
#else
#define STAGE_ASYNC 0
#endif

__device__ inline void stage_weights(const bf16* __restrict__ gsrc,
                                     bf16* lds_dst, int nelem) {
#if STAGE_TDM
    if (threadIdx.x < 32) {   // one wave owns the TDM op (EXEC ignored by TDM)
        const unsigned lds_off = (unsigned)(unsigned long long)(void*)lds_dst;
        const unsigned long long ga = (unsigned long long)(const void*)gsrc;
        u32x4 g0;
        g0[0] = 1u;                                           // count=1, user D#
        g0[1] = lds_off;                                      // lds_addr
        g0[2] = (unsigned)ga;                                 // global_addr[31:0]
        g0[3] = (unsigned)((ga >> 32) & 0x01FFFFFFu) | (2u << 30); // addr[56:32] | type=2
        i32x8 g1;
        g1[0] = 0x00010000;                                   // data_size=1 (2 bytes)
        g1[1] = (int)(((unsigned)nelem & 0xFFFFu) << 16);     // tensor_dim0[15:0]
        g1[2] = (int)((((unsigned)nelem >> 16) & 0xFFFFu) | (1u << 16)); // dim0 hi | tensor_dim1=1
        g1[3] = (int)(((unsigned)nelem & 0xFFFFu) << 16);     // tile_dim0
        g1[4] = 1;                                            // tile_dim1=1, tile_dim2=0
        g1[5] = nelem;                                        // tensor_dim0_stride lo32
        g1[6] = 0;
        g1[7] = 0;
        i32x4 gz; gz[0] = gz[1] = gz[2] = gz[3] = 0;
#if __clang_major__ >= 23
        i32x8 gz8; for (int j = 0; j < 8; ++j) gz8[j] = 0;
        __builtin_amdgcn_tensor_load_to_lds(g0, g1, gz, gz, gz8, 0);
#else
        __builtin_amdgcn_tensor_load_to_lds(g0, g1, gz, gz, 0);
#endif
        __builtin_amdgcn_s_wait_tensorcnt(0);
    }
#elif STAGE_ASYNC
    const uint4* s = (const uint4*)gsrc;
    uint4* d = (uint4*)lds_dst;
    for (int i = threadIdx.x; i < (nelem >> 3); i += 256) {
        __builtin_amdgcn_global_load_async_to_lds_b128(
            (__attribute__((address_space(1))) void*)(unsigned long long)(const void*)(s + i),
            (__attribute__((address_space(3))) void*)(unsigned)(unsigned long long)(void*)(d + i),
            0, 0);
    }
    __builtin_amdgcn_s_wait_asynccnt(0);
#else
    const uint4* s = (const uint4*)gsrc;
    uint4* d = (uint4*)lds_dst;
    for (int i = threadIdx.x; i < (nelem >> 3); i += 256) d[i] = s[i];
#endif
}

// ---------------------------------------------------------------------------
// Stats zeroing (7 conv layers x [sum[64], sumsq[64]])
// ---------------------------------------------------------------------------
__global__ void zero_stats_kernel(float* __restrict__ p) {
    int i = blockIdx.x * 256 + threadIdx.x;
    if (i < 7 * 128) p[i] = 0.0f;
}

// ---------------------------------------------------------------------------
// Instance norm (unbiased, no eps) + pack [x, norm, 0, 0] -> NHWC8 bf16
// One block per (n, c); x is NCHW fp32.
// ---------------------------------------------------------------------------
__global__ __launch_bounds__(256) void instnorm_pack_kernel(
    const float* __restrict__ x, bf16* __restrict__ inp8)
{
    const int n = blockIdx.x / 3, c = blockIdx.x % 3;
    const float* xp = x + ((size_t)n * 3 + c) * 65536;
    float s = 0.0f, q = 0.0f;
    for (int i = threadIdx.x; i < 65536; i += 256) {
        float v = xp[i]; s += v; q += v * v;
    }
    __shared__ float rs[256], rq[256];
    rs[threadIdx.x] = s; rq[threadIdx.x] = q;
    __syncthreads();
    for (int off = 128; off > 0; off >>= 1) {
        if (threadIdx.x < off) {
            rs[threadIdx.x] += rs[threadIdx.x + off];
            rq[threadIdx.x] += rq[threadIdx.x + off];
        }
        __syncthreads();
    }
    const float mean = rs[0] * (1.0f / 65536.0f);
    const float var  = (rq[0] - 65536.0f * mean * mean) * (1.0f / 65535.0f);
    const float inv  = rsqrtf(var);
    bf16* op = inp8 + (size_t)n * 65536 * 8;
    for (int i = threadIdx.x; i < 65536; i += 256) {
        float v = xp[i];
        op[(size_t)i * 8 + c]     = (bf16)v;
        op[(size_t)i * 8 + 3 + c] = (bf16)((v - mean) * inv);
        if (c == 0) {
            op[(size_t)i * 8 + 6] = (bf16)0.0f;
            op[(size_t)i * 8 + 7] = (bf16)0.0f;
        }
    }
}

// ---------------------------------------------------------------------------
// Weight transforms (fp32 OIHW -> bf16 WMMA-friendly layouts)
// ---------------------------------------------------------------------------
__global__ void wprep_res_kernel(const float* __restrict__ bw, bf16* __restrict__ wT) {
    int idx = blockIdx.x * 256 + threadIdx.x;           // 6*9*64*64 = 221184
    if (idx >= 6 * 9 * 64 * 64) return;
    int cin = idx & 63, o = (idx >> 6) & 63, lt = idx >> 12;
    int l = lt / 9, t = lt % 9;
    int ky = t / 3, kx = t % 3;
    size_t src = ((((size_t)l * 64 + o) * 64 + cin) * 3 + ky) * 3 + kx;  // [3][2] flat == l
    wT[idx] = (bf16)bw[src];
}

__global__ void wprep1_kernel(const float* __restrict__ w1, bf16* __restrict__ wB1) {
    int idx = blockIdx.x * 256 + threadIdx.x;           // 64*96 = 6144
    if (idx >= 64 * 96) return;
    int k = idx % 96, o = idx / 96;
    int t = k >> 3, c = k & 7;
    float v = 0.0f;
    if (t < 9 && c < 6) {
        int ky = t / 3, kx = t % 3;
        v = w1[(((size_t)o * 6 + c) * 3 + ky) * 3 + kx];
    }
    wB1[idx] = (bf16)v;
}

__global__ void wprep2_kernel(const float* __restrict__ w2, bf16* __restrict__ wB2) {
    int idx = blockIdx.x * 256 + threadIdx.x;           // 16*576 = 9216
    if (idx >= 16 * 576) return;
    int k = idx % 576, o = idx / 576;
    float v = 0.0f;
    if (o < 3) {
        int t = k / 64, c = k % 64;
        int ky = t / 3, kx = t % 3;
        v = w2[(((size_t)o * 64 + c) * 3 + ky) * 3 + kx];
    }
    wB2[idx] = (bf16)v;
}

// ---------------------------------------------------------------------------
// Main 64->64 3x3 conv, implicit GEMM via v_wmma_f32_16x16x32_bf16.
// Wave computes 32 pixels x 64 couts; weights staged in LDS; fused BN stats.
// ---------------------------------------------------------------------------
__global__ __launch_bounds__(256) void conv_wmma_kernel(
    const bf16* __restrict__ actIn,   // [NPIX][64] NHWC bf16
    const bf16* __restrict__ wT,      // [9][64][64] tap,cout,cin bf16
    bf16* __restrict__ convOut,       // [NPIX][64]
    float* __restrict__ stats)        // [128] sum, sumsq
{
    __shared__ __align__(16) bf16 wl[9 * 64 * 64];
    __shared__ float s_sum[64], s_sq[64];
    stage_weights(wT, wl, 9 * 64 * 64);
    if (threadIdx.x < 64) { s_sum[threadIdx.x] = 0.0f; s_sq[threadIdx.x] = 0.0f; }
    __syncthreads();

    const int lane = threadIdx.x & 31, half = lane >> 4, m = lane & 15;
    const int gw = blockIdx.x * 8 + (threadIdx.x >> 5);
    const int stride = gridDim.x * 8;
    const int nT = NB * HH * (WW / 32);   // 16384 pair-tiles

    for (int t = gw; t < nT; t += stride) {
        const int row = t >> 3;            // n*H + h
        const int w0  = (t & 7) << 5;      // 32-pixel tile start
        const int h   = row & 255;

        v8f acc[8] = {};                   // [mt][nc]
        #pragma unroll
        for (int dy = -1; dy <= 1; ++dy) {
            const int ih = h + dy;
            if (ih < 0 || ih >= HH) continue;
            const size_t rb = (size_t)(row + dy) * WW;
            #pragma unroll
            for (int dx = -1; dx <= 1; ++dx) {
                const int tap = (dy + 1) * 3 + (dx + 1);
                #pragma unroll
                for (int kc = 0; kc < 2; ++kc) {
                    Frag A[2];
                    #pragma unroll
                    for (int mt = 0; mt < 2; ++mt) {
                        int iw = w0 + mt * 16 + m + dx;
                        bool oob = (iw < 0) || (iw >= WW);
                        int iwc = iw < 0 ? 0 : (iw >= WW ? WW - 1 : iw);
                        const bf16* ap = actIn + (rb + iwc) * 64 + kc * 32 + half * 8;
                        uint4 a0 = *(const uint4*)ap;         // K = half*8 + 0..7
                        uint4 a1 = *(const uint4*)(ap + 16);  // K = 16 + half*8 + 0..7
                        if (oob) { a0 = uzero(); a1 = uzero(); }
                        A[mt].u[0] = a0; A[mt].u[1] = a1;
                    }
                    #pragma unroll
                    for (int nc = 0; nc < 4; ++nc) {
                        Frag B;  // lane(half,n): K = half*8 + 0..7, 16 + half*8 + 0..7 of cout row n
                        const bf16* bp = wl + tap * 4096 + (nc * 16 + m) * 64 + kc * 32 + half * 8;
                        B.u[0] = *(const uint4*)bp;
                        B.u[1] = *(const uint4*)(bp + 16);
                        acc[0 * 4 + nc] = __builtin_amdgcn_wmma_f32_16x16x32_bf16(
                            false, A[0].v, false, B.v, (short)0, acc[0 * 4 + nc], false, false);
                        acc[1 * 4 + nc] = __builtin_amdgcn_wmma_f32_16x16x32_bf16(
                            false, A[1].v, false, B.v, (short)0, acc[1 * 4 + nc], false, false);
                    }
                }
            }
        }
        // epilogue: store bf16 conv output + per-channel partial stats
        #pragma unroll
        for (int mt = 0; mt < 2; ++mt) {
            const size_t pixBase = (size_t)row * WW + w0 + mt * 16 + half * 8;
            #pragma unroll
            for (int nc = 0; nc < 4; ++nc) {
                float s = 0.0f, q = 0.0f;
                #pragma unroll
                for (int r = 0; r < 8; ++r) {
                    float v = acc[mt * 4 + nc][r];
                    s += v; q += v * v;
                    convOut[(pixBase + r) * 64 + nc * 16 + m] = (bf16)v;
                }
                atomicAdd(&s_sum[nc * 16 + m], s);
                atomicAdd(&s_sq [nc * 16 + m], q);
            }
        }
    }
    __syncthreads();
    if (threadIdx.x < 64) {
        atomicAdd(&stats[threadIdx.x],      s_sum[threadIdx.x]);
        atomicAdd(&stats[64 + threadIdx.x], s_sq [threadIdx.x]);
    }
}

// ---------------------------------------------------------------------------
// First conv (6ch padded to 8) via WMMA: K = tap*8 + c, 96 total (3 kchunks).
// Each 16B A-subchunk == one pixel's 8 packed channels.
// ---------------------------------------------------------------------------
__device__ inline uint4 ld_tap_chunk(const bf16* __restrict__ inp8,
                                     int row, int h, int wm, int t) {
    if (t > 8) return uzero();
    const int dy = t / 3 - 1, dx = t % 3 - 1;
    const int ih = h + dy;
    const int iw = wm + dx;
    if (ih < 0 || ih >= HH || iw < 0 || iw >= WW) return uzero();
    return *(const uint4*)(inp8 + ((size_t)(row + dy) * WW + iw) * 8);
}

__global__ __launch_bounds__(256) void conv1_wmma_kernel(
    const bf16* __restrict__ inp8,    // [NPIX][8]
    const bf16* __restrict__ wB1,     // [64][96]
    bf16* __restrict__ convOut,
    float* __restrict__ stats)
{
    __shared__ __align__(16) bf16 wl[64 * 96];
    __shared__ float s_sum[64], s_sq[64];
    stage_weights(wB1, wl, 64 * 96);
    if (threadIdx.x < 64) { s_sum[threadIdx.x] = 0.0f; s_sq[threadIdx.x] = 0.0f; }
    __syncthreads();

    const int lane = threadIdx.x & 31, half = lane >> 4, m = lane & 15;
    const int gw = blockIdx.x * 8 + (threadIdx.x >> 5);
    const int stride = gridDim.x * 8;

    for (int t = gw; t < NB * HH * (WW / 16); t += stride) {   // 32768 tiles
        const int row = t >> 4, w0 = (t & 15) << 4, h = row & 255;
        v8f acc[4] = {};
        #pragma unroll
        for (int kc = 0; kc < 3; ++kc) {
            Frag A;
            A.u[0] = ld_tap_chunk(inp8, row, h, w0 + m, kc * 4 + half);
            A.u[1] = ld_tap_chunk(inp8, row, h, w0 + m, kc * 4 + 2 + half);
            #pragma unroll
            for (int nc = 0; nc < 4; ++nc) {
                Frag B;
                const bf16* bp = wl + (nc * 16 + m) * 96 + kc * 32 + half * 8;
                B.u[0] = *(const uint4*)bp;
                B.u[1] = *(const uint4*)(bp + 16);
                acc[nc] = __builtin_amdgcn_wmma_f32_16x16x32_bf16(
                    false, A.v, false, B.v, (short)0, acc[nc], false, false);
            }
        }
        const size_t pixBase = (size_t)row * WW + w0 + half * 8;
        #pragma unroll
        for (int nc = 0; nc < 4; ++nc) {
            float s = 0.0f, q = 0.0f;
            #pragma unroll
            for (int r = 0; r < 8; ++r) {
                float v = acc[nc][r];
                s += v; q += v * v;
                convOut[(pixBase + r) * 64 + nc * 16 + m] = (bf16)v;
            }
            atomicAdd(&s_sum[nc * 16 + m], s);
            atomicAdd(&s_sq [nc * 16 + m], q);
        }
    }
    __syncthreads();
    if (threadIdx.x < 64) {
        atomicAdd(&stats[threadIdx.x],      s_sum[threadIdx.x]);
        atomicAdd(&stats[64 + threadIdx.x], s_sq [threadIdx.x]);
    }
}

// ---------------------------------------------------------------------------
// BN (training-mode, biased var) + CELU (+ optional residual), bf16 in/out.
// ---------------------------------------------------------------------------
__global__ __launch_bounds__(256) void bn_apply_kernel(
    const bf16* __restrict__ conv, const float* __restrict__ stats,
    const float* __restrict__ gamma, const float* __restrict__ beta,
    const bf16* __restrict__ skip, bf16* __restrict__ outAct)
{
    __shared__ float sc[64], sh[64];
    if (threadIdx.x < 64) {
        const int c = threadIdx.x;
        const float invM = 1.0f / (float)NPIX;
        float mean = stats[c] * invM;
        float var  = stats[64 + c] * invM - mean * mean;
        float s = gamma[c] * rsqrtf(var + 1e-5f);
        sc[c] = s;
        sh[c] = beta[c] - mean * s;
    }
    __syncthreads();
    const size_t base = ((size_t)blockIdx.x * 256 + threadIdx.x) * 8;
    if (base >= (size_t)NPIX * 64) return;
    const int cb = (int)(base & 63);
    uint4 u = *(const uint4*)(conv + base);
    const bf16* cv = (const bf16*)&u;
    union { bf16 b[8]; uint4 u4; } O;
    if (skip) {
        uint4 su = *(const uint4*)(skip + base);
        const bf16* sv = (const bf16*)&su;
        #pragma unroll
        for (int j = 0; j < 8; ++j) {
            float y = (float)cv[j] * sc[cb + j] + sh[cb + j];
            y = y > 0.0f ? y : (__expf(y) - 1.0f);
            O.b[j] = (bf16)(y + (float)sv[j]);
        }
    } else {
        #pragma unroll
        for (int j = 0; j < 8; ++j) {
            float y = (float)cv[j] * sc[cb + j] + sh[cb + j];
            y = y > 0.0f ? y : (__expf(y) - 1.0f);
            O.b[j] = (bf16)y;
        }
    }
    *(uint4*)(outAct + base) = O.u4;
}

// ---------------------------------------------------------------------------
// Last conv 64->3 (cout padded to 16) via WMMA; fused bias + CELU + (+x),
// writes fp32 NCHW output.
// ---------------------------------------------------------------------------
__global__ __launch_bounds__(256) void conv2_wmma_kernel(
    const bf16* __restrict__ actIn,   // [NPIX][64]
    const bf16* __restrict__ wB2,     // [16][576]
    const float* __restrict__ b2,     // [3]
    const float* __restrict__ x,      // NCHW fp32
    float* __restrict__ out)          // NCHW fp32
{
    __shared__ __align__(16) bf16 wl[16 * 576];
    stage_weights(wB2, wl, 16 * 576);
    __syncthreads();
    const int lane = threadIdx.x & 31, half = lane >> 4, m = lane & 15;
    const int gw = blockIdx.x * 8 + (threadIdx.x >> 5);
    const int stride = gridDim.x * 8;

    for (int t = gw; t < NB * HH * (WW / 16); t += stride) {
        const int row = t >> 4, w0 = (t & 15) << 4, h = row & 255, n = row >> 8;
        v8f acc = {};
        #pragma unroll
        for (int kc = 0; kc < 18; ++kc) {        // K = tap*64 + c
            const int tap = kc >> 1;
            const int dy = tap / 3 - 1, dx = tap % 3 - 1;
            const int ih = h + dy;
            if (ih < 0 || ih >= HH) continue;
            int iw = w0 + m + dx;
            bool oob = (iw < 0) || (iw >= WW);
            int iwc = iw < 0 ? 0 : (iw >= WW ? WW - 1 : iw);
            const bf16* ap = actIn + ((size_t)(row + dy) * WW + iwc) * 64
                             + (kc & 1) * 32 + half * 8;
            uint4 a0 = *(const uint4*)ap;
            uint4 a1 = *(const uint4*)(ap + 16);
            if (oob) { a0 = uzero(); a1 = uzero(); }
            Frag A; A.u[0] = a0; A.u[1] = a1;
            Frag B;
            const bf16* bp = wl + m * 576 + kc * 32 + half * 8;
            B.u[0] = *(const uint4*)bp;
            B.u[1] = *(const uint4*)(bp + 16);
            acc = __builtin_amdgcn_wmma_f32_16x16x32_bf16(
                false, A.v, false, B.v, (short)0, acc, false, false);
        }
        if (m < 3) {
            const float bias = b2[m];
            const size_t obase = ((size_t)n * 3 + m) * 65536
                               + (size_t)h * 256 + w0 + half * 8;
            #pragma unroll
            for (int r = 0; r < 8; ++r) {
                float y = acc[r] + bias;
                y = y > 0.0f ? y : (__expf(y) - 1.0f);
                out[obase + r] = y + x[obase + r];
            }
        }
    }
}

// ---------------------------------------------------------------------------
extern "C" void kernel_launch(void* const* d_in, const int* in_sizes, int n_in,
                              void* d_out, int out_size, void* d_ws, size_t ws_size,
                              hipStream_t stream) {
    (void)in_sizes; (void)n_in; (void)out_size; (void)ws_size;
    const float* x   = (const float*)d_in[0];
    const float* w1  = (const float*)d_in[1];
    // b1 (d_in[2]) cancels under training-mode BN
    const float* g1  = (const float*)d_in[3];
    const float* be1 = (const float*)d_in[4];
    const float* bw  = (const float*)d_in[5];
    // bb (d_in[6]) cancels under training-mode BN
    const float* bg  = (const float*)d_in[7];
    const float* bbe = (const float*)d_in[8];
    const float* w2  = (const float*)d_in[9];
    const float* b2  = (const float*)d_in[10];
    float* out = (float*)d_out;

    char* ws = (char*)d_ws;
    bf16*  inp8  = (bf16*)(ws);                              //  8,388,608 B
    bf16*  convT = (bf16*)(ws + 8388608);                    // 67,108,864 B
    bf16*  actA  = (bf16*)(ws + 8388608 + 67108864);         // 67,108,864 B
    bf16*  actB  = (bf16*)(ws + 8388608 + 2 * 67108864);     // 67,108,864 B
    bf16*  wT    = (bf16*)(ws + 8388608 + 3 * (size_t)67108864);          // 442,368 B
    bf16*  wB1   = (bf16*)(ws + 8388608 + 3 * (size_t)67108864 + 442368); //  12,288 B
    bf16*  wB2   = (bf16*)(ws + 8388608 + 3 * (size_t)67108864 + 454656); //  18,432 B
    float* stats = (float*)(ws + 8388608 + 3 * (size_t)67108864 + 473088);//   3,584 B

    zero_stats_kernel<<<4, 256, 0, stream>>>(stats);
    wprep_res_kernel<<<864, 256, 0, stream>>>(bw, wT);
    wprep1_kernel<<<24, 256, 0, stream>>>(w1, wB1);
    wprep2_kernel<<<36, 256, 0, stream>>>(w2, wB2);
    instnorm_pack_kernel<<<24, 256, 0, stream>>>(x, inp8);

    conv1_wmma_kernel<<<1024, 256, 0, stream>>>(inp8, wB1, convT, stats);
    bn_apply_kernel<<<16384, 256, 0, stream>>>(convT, stats, g1, be1,
                                               (const bf16*)nullptr, actA);
    for (int i = 0; i < 3; ++i) {
        conv_wmma_kernel<<<1024, 256, 0, stream>>>(
            actA, wT + (size_t)(i * 2) * 36864, convT, stats + (1 + i * 2) * 128);
        bn_apply_kernel<<<16384, 256, 0, stream>>>(
            convT, stats + (1 + i * 2) * 128, bg + (i * 2) * 64, bbe + (i * 2) * 64,
            (const bf16*)nullptr, actB);
        conv_wmma_kernel<<<1024, 256, 0, stream>>>(
            actB, wT + (size_t)(i * 2 + 1) * 36864, convT, stats + (2 + i * 2) * 128);
        bn_apply_kernel<<<16384, 256, 0, stream>>>(
            convT, stats + (2 + i * 2) * 128, bg + (i * 2 + 1) * 64, bbe + (i * 2 + 1) * 64,
            actA, actA);
    }
    conv2_wmma_kernel<<<1024, 256, 0, stream>>>(actA, wB2, b2, x, out);
}
